// UniversalTransformerEncoder_6073083756959
// MI455X (gfx1250) — compile-verified, matrix operand order checked by reading, not run on hardware
//
#include <hip/hip_runtime.h>

// ---------------- problem constants ----------------
static constexpr int Bn   = 2;
static constexpr int Sn   = 1024;
static constexpr int Dn   = 1024;
static constexpr int Hn   = 16;
static constexpr int DHn  = 64;
static constexpr int Fn   = 4096;
static constexpr int STEPS = 6;
static constexpr int Mn   = Bn * Sn;        // 2048 tokens
static constexpr int BHn  = Bn * Hn;        // 32 attention pairs
static constexpr int CHn  = 8;              // pairs per attention chunk

// LDS panel: 128 rows x 32 bf16, padded to 40 elements/row (80B stride:
// 20-dword stride, gcd(20,64)=4 -> 16 distinct 4-bank groups -> conflict-free b128 reads)
static constexpr int KC     = 32;
static constexpr int LDSPAD = 40;

// async global->LDS path (gfx1250) if the toolchain exposes it
#if defined(__has_builtin)
#  if __has_builtin(__builtin_amdgcn_global_load_async_to_lds_b128) && \
      __has_builtin(__builtin_amdgcn_s_wait_asynccnt)
#    define UT_ASYNC 1
#  endif
#endif

// ---------------- WMMA types ----------------
typedef __attribute__((ext_vector_type(16))) __bf16        v16bf;
typedef __attribute__((ext_vector_type(8)))  float         v8f;
typedef __attribute__((ext_vector_type(4)))  unsigned int  v4u;
typedef __attribute__((ext_vector_type(4)))  int           v4i;

union ABFrag { v16bf bf; v4u u[2]; };

__device__ __forceinline__ unsigned short f2bf(float f) {
    unsigned int u = __float_as_uint(f);
    unsigned int r = u + 0x7FFFu + ((u >> 16) & 1u);   // round-to-nearest-even
    return (unsigned short)(r >> 16);
}

// ---------------- direct-from-global WMMA tile loop (small-K attention GEMM) ----
__device__ __forceinline__ v8f wmma_loop(const unsigned short* __restrict__ A, size_t lda,
                                         const unsigned short* __restrict__ Bt, size_t ldb,
                                         int K) {
    int lane = threadIdx.x & 31;
    int half = lane >> 4;
    int r    = lane & 15;
    v8f acc = {0.f, 0.f, 0.f, 0.f, 0.f, 0.f, 0.f, 0.f};
    const unsigned short* arow = A  + (size_t)r * lda;
    const unsigned short* brow = Bt + (size_t)r * ldb;
    for (int k0 = 0; k0 < K; k0 += 32) {
        int klo = k0 + half * 8;
        int khi = k0 + 16 + half * 8;
        ABFrag a, b;
        a.u[0] = *reinterpret_cast<const v4u*>(arow + klo);
        a.u[1] = *reinterpret_cast<const v4u*>(arow + khi);
        b.u[0] = *reinterpret_cast<const v4u*>(brow + klo);
        b.u[1] = *reinterpret_cast<const v4u*>(brow + khi);
        __builtin_prefetch(brow + k0 + 128, 0, 1);
        acc = __builtin_amdgcn_wmma_f32_16x16x32_bf16(false, a.bf, false, b.bf,
                                                      (short)0, acc, false, false);
    }
    return acc;
}

// store 16x16 fp32 C/D-layout tile with optional bias / residual / relu / bf16 out
__device__ __forceinline__ void store_tile(v8f acc, float* Cf, unsigned short* Cb, size_t ldc,
                                           const float* bias, const float* residual, int relu) {
    int lane = threadIdx.x & 31;
    int half = lane >> 4;
    int nl   = lane & 15;
    float bv = bias ? bias[nl] : 0.f;
    #pragma unroll
    for (int i = 0; i < 8; ++i) {
        size_t off = (size_t)(i + half * 8) * ldc + nl;
        float v = acc[i] + bv;
        if (residual) v += residual[off];
        if (relu) v = fmaxf(v, 0.f);
        if (Cf) Cf[off] = v;
        if (Cb) Cb[off] = f2bf(v);
    }
}

// ---------------- LDS-tiled GEMM (linear layers) ----------------
// stage one 128 x 32 bf16 panel into an LDS buffer (per-thread: 2 x 16B chunks)
__device__ __forceinline__ void stage_panel(unsigned short* __restrict__ buf,
                                            const unsigned short* __restrict__ src,
                                            size_t ld, int k0) {
    #pragma unroll
    for (int j = 0; j < 2; ++j) {
        int c   = threadIdx.x + j * 256;   // 0..511
        int row = c >> 2;                  // 0..127
        int cc  = (c & 3) * 8;             // 0,8,16,24 (bf16 elements)
        const unsigned short* g = src + (size_t)row * ld + k0 + cc;
        unsigned short* l = buf + row * LDSPAD + cc;
#ifdef UT_ASYNC
        __builtin_amdgcn_global_load_async_to_lds_b128(
            (__attribute__((address_space(1))) v4i*)g,
            (__attribute__((address_space(3))) v4i*)l, 0, 0);
#else
        *reinterpret_cast<v4u*>(l) = *reinterpret_cast<const v4u*>(g);
#endif
    }
}

// load one 16x32 fragment from an LDS panel (tile row 0 at `base`)
__device__ __forceinline__ v16bf ld_frag_lds(const unsigned short* __restrict__ base) {
    int lane = threadIdx.x & 31;
    int half = lane >> 4;
    int r    = lane & 15;
    const unsigned short* p = base + r * LDSPAD;
    ABFrag f;
    f.u[0] = *reinterpret_cast<const v4u*>(p + half * 8);
    f.u[1] = *reinterpret_cast<const v4u*>(p + 16 + half * 8);
    return f.bf;
}

// block = 8 waves -> 128x128 output tile; wave = 64x32 (4x2 accumulators);
// grid: x = N/128, y = M/128
__global__ __launch_bounds__(256) void gemm_tiled(const unsigned short* __restrict__ A, int lda,
                                                  const unsigned short* __restrict__ Bt, int ldb,
                                                  int K,
                                                  float* Cf, unsigned short* Cb, int ldc,
                                                  const float* bias, const float* residual,
                                                  int relu) {
    __shared__ __align__(16) unsigned short ldsA[2][128 * LDSPAD];
    __shared__ __align__(16) unsigned short ldsB[2][128 * LDSPAD];
    const int m0 = blockIdx.y * 128;
    const int n0 = blockIdx.x * 128;
    const unsigned short* Ab = A  + (size_t)m0 * lda;
    const unsigned short* Bb = Bt + (size_t)n0 * ldb;
    const int wid   = threadIdx.x >> 5;
    const int waveM = wid >> 2;      // 0..1
    const int waveN = wid & 3;       // 0..3

    v8f zc = {0.f, 0.f, 0.f, 0.f, 0.f, 0.f, 0.f, 0.f};
    v8f acc[4][2];
    #pragma unroll
    for (int mt = 0; mt < 4; ++mt)
        #pragma unroll
        for (int nt = 0; nt < 2; ++nt) acc[mt][nt] = zc;

    const int nk = K >> 5;
    stage_panel(ldsA[0], Ab, lda, 0);
    stage_panel(ldsB[0], Bb, ldb, 0);

    for (int i = 0; i < nk; ++i) {
        const int cur = i & 1;
        if (i + 1 < nk) {
            stage_panel(ldsA[cur ^ 1], Ab, lda, (i + 1) << 5);
            stage_panel(ldsB[cur ^ 1], Bb, ldb, (i + 1) << 5);
#ifdef UT_ASYNC
            __builtin_amdgcn_s_wait_asynccnt(4);   // allow the 4 just-issued, require prior stage done
#endif
        }
#ifdef UT_ASYNC
        else {
            __builtin_amdgcn_s_wait_asynccnt(0);
        }
#endif
        __syncthreads();

        v16bf af[4], bfr[2];
        #pragma unroll
        for (int mt = 0; mt < 4; ++mt)
            af[mt] = ld_frag_lds(&ldsA[cur][(waveM * 64 + mt * 16) * LDSPAD]);
        #pragma unroll
        for (int nt = 0; nt < 2; ++nt)
            bfr[nt] = ld_frag_lds(&ldsB[cur][(waveN * 32 + nt * 16) * LDSPAD]);
        #pragma unroll
        for (int mt = 0; mt < 4; ++mt)
            #pragma unroll
            for (int nt = 0; nt < 2; ++nt)
                acc[mt][nt] = __builtin_amdgcn_wmma_f32_16x16x32_bf16(
                    false, af[mt], false, bfr[nt], (short)0, acc[mt][nt], false, false);
        __syncthreads();
    }

    #pragma unroll
    for (int mt = 0; mt < 4; ++mt)
        #pragma unroll
        for (int nt = 0; nt < 2; ++nt) {
            int nn = n0 + waveN * 32 + nt * 16;
            size_t cbase = (size_t)(m0 + waveM * 64 + mt * 16) * ldc + nn;
            store_tile(acc[mt][nt],
                       Cf ? Cf + cbase : nullptr,
                       Cb ? Cb + cbase : nullptr, ldc,
                       bias ? bias + nn : nullptr,
                       residual ? residual + cbase : nullptr, relu);
        }
}

// ---------------- attention GEMMs ----------------
// logits[z, q, k] = Q_h[q,:] . K_h[k,:]   (per (b,h) pair; K dim = 64)
__global__ __launch_bounds__(256) void gemm_qk(const unsigned short* __restrict__ qkv,
                                               int pairBase, float* __restrict__ logits) {
    int z = blockIdx.z;
    int pair = pairBase + z;
    int b = pair / Hn, h = pair % Hn;
    const int ntiles = Sn >> 4;   // 64
    int tile = blockIdx.x * 8 + (threadIdx.x >> 5);
    if (tile >= ntiles * ntiles) return;
    int tm = tile / ntiles, tn = tile % ntiles;
    const unsigned short* Ab = qkv + (size_t)b * Sn * 3 * Dn + (size_t)h * DHn;
    const unsigned short* Kb = qkv + (size_t)b * Sn * 3 * Dn + Dn + (size_t)h * DHn;
    v8f acc = wmma_loop(Ab + (size_t)tm * 16 * (3 * Dn), 3 * Dn,
                        Kb + (size_t)tn * 16 * (3 * Dn), 3 * Dn, DHn);
    float* C = logits + (size_t)z * Sn * Sn + (size_t)tm * 16 * Sn + (size_t)tn * 16;
    store_tile(acc, C, nullptr, Sn, nullptr, nullptr, 0);
}

// out[b, q, h*64+n] = attn[z,q,:] . Vt[pair, n, :]; wave computes 16 x 64 (whole head)
__global__ __launch_bounds__(256) void gemm_av(const unsigned short* __restrict__ attn,
                                               const unsigned short* __restrict__ vt,
                                               int pairBase, unsigned short* __restrict__ outbf) {
    int z = blockIdx.z;
    int pair = pairBase + z;
    int b = pair / Hn, h = pair % Hn;
    int tm = blockIdx.x * 8 + (threadIdx.x >> 5);   // 0..63 (grid.x = 8)
    int lane = threadIdx.x & 31;
    int half = lane >> 4, r = lane & 15;
    const unsigned short* arow  = attn + (size_t)z * Sn * Sn + ((size_t)tm * 16 + r) * Sn;
    const unsigned short* vbase = vt + (size_t)pair * DHn * Sn;
    v8f acc[4];
    v8f zc = {0.f, 0.f, 0.f, 0.f, 0.f, 0.f, 0.f, 0.f};
    #pragma unroll
    for (int nt = 0; nt < 4; ++nt) acc[nt] = zc;
    for (int k0 = 0; k0 < Sn; k0 += 32) {
        ABFrag a;
        a.u[0] = *reinterpret_cast<const v4u*>(arow + k0 + half * 8);
        a.u[1] = *reinterpret_cast<const v4u*>(arow + k0 + 16 + half * 8);
        #pragma unroll
        for (int nt = 0; nt < 4; ++nt) {
            const unsigned short* brow = vbase + ((size_t)nt * 16 + r) * Sn;
            ABFrag bb;
            bb.u[0] = *reinterpret_cast<const v4u*>(brow + k0 + half * 8);
            bb.u[1] = *reinterpret_cast<const v4u*>(brow + k0 + 16 + half * 8);
            acc[nt] = __builtin_amdgcn_wmma_f32_16x16x32_bf16(
                false, a.bf, false, bb.bf, (short)0, acc[nt], false, false);
        }
    }
    #pragma unroll
    for (int nt = 0; nt < 4; ++nt) {
        unsigned short* C = outbf + (size_t)b * Sn * Dn + (size_t)h * DHn +
                            (size_t)tm * 16 * Dn + (size_t)nt * 16;
        store_tile(acc[nt], nullptr, C, Dn, nullptr, nullptr, 0);
    }
}

// ---------------- elementwise / prep kernels ----------------
__global__ __launch_bounds__(256) void transpose_w(const float* __restrict__ src,
                                                   unsigned short* __restrict__ dst,
                                                   int K, int N) {
    size_t idx = (size_t)blockIdx.x * 256 + threadIdx.x;
    if (idx >= (size_t)K * N) return;
    int k = (int)(idx % K);
    int n = (int)(idx / K);
    dst[idx] = f2bf(src[(size_t)k * N + n]);
}

__global__ __launch_bounds__(256) void transpose_v(const unsigned short* __restrict__ qkv,
                                                   unsigned short* __restrict__ vt) {
    size_t idx = (size_t)blockIdx.x * 256 + threadIdx.x;   // B*H*DH*S = 2M
    int s = (int)(idx & (Sn - 1));
    int n = (int)((idx >> 10) & (DHn - 1));
    int pair = (int)(idx >> 16);
    int b = pair / Hn, h = pair % Hn;
    vt[idx] = qkv[((size_t)b * Sn + s) * (3 * Dn) + 2 * Dn + (size_t)h * DHn + n];
}

__global__ __launch_bounds__(256) void pos_step(const float* __restrict__ src,
                                                const float* __restrict__ emb,
                                                float* __restrict__ st) {
    size_t idx = (size_t)blockIdx.x * 256 + threadIdx.x;   // B*S*D = 2M
    int d = (int)(idx & (Dn - 1));
    int s = (int)((idx >> 10) & (Sn - 1));
    const float LOG_INC = 0.0180241506f;                   // ln(1e4)/511
    float sig;
    if (d < 512) sig = sinf((float)s * expf(-(float)d * LOG_INC));
    else         sig = cosf((float)s * expf(-(float)(d - 512) * LOG_INC));
    st[idx] = src[idx] + sig + emb[d];
}

__global__ __launch_bounds__(256) void act_update(const float* __restrict__ st,
                                                  const float* __restrict__ hw,
                                                  const float* __restrict__ hb,
                                                  float* halting, float* remainders,
                                                  float* nupd, float* upw) {
    int wave = threadIdx.x >> 5, lane = threadIdx.x & 31;
    int t = blockIdx.x * 8 + wave;
    const float* xr = st + (size_t)t * Dn;
    float d = 0.f;
    for (int i = lane; i < Dn; i += 32) d += xr[i] * hw[i];
    for (int o = 16; o; o >>= 1) d += __shfl_down(d, o, 32);
    if (lane == 0) {
        float p  = 1.f / (1.f + expf(-(d + hb[0])));
        float h  = halting[t];
        float sr = (h < 1.0f) ? 1.f : 0.f;
        float hp = h + p * sr;
        float nh  = (hp >  0.99f) ? sr : 0.f;
        float sr2 = (hp <= 0.99f) ? sr : 0.f;
        h = h + p * sr2;
        float rem = remainders[t] + nh * (1.f - h);
        h = h + nh * rem;
        nupd[t] += sr2 + nh;
        upw[t] = p * sr2 + nh * rem;
        halting[t] = h;
        remainders[t] = rem;
    }
}

__global__ __launch_bounds__(256) void ln_kernel(const float* __restrict__ x,
                                                 const float* __restrict__ sc,
                                                 const float* __restrict__ bi,
                                                 unsigned short* ybf, float* yf) {
    __shared__ float red[256];
    int row = blockIdx.x;
    const float* xr = x + (size_t)row * Dn;
    float vals[4];
    float s = 0.f;
    #pragma unroll
    for (int i = 0; i < 4; ++i) { vals[i] = xr[threadIdx.x + i * 256]; s += vals[i]; }
    red[threadIdx.x] = s; __syncthreads();
    for (int o = 128; o; o >>= 1) { if (threadIdx.x < o) red[threadIdx.x] += red[threadIdx.x + o]; __syncthreads(); }
    float mean = red[0] * (1.f / Dn); __syncthreads();
    float q = 0.f;
    #pragma unroll
    for (int i = 0; i < 4; ++i) { float dd = vals[i] - mean; q += dd * dd; }
    red[threadIdx.x] = q; __syncthreads();
    for (int o = 128; o; o >>= 1) { if (threadIdx.x < o) red[threadIdx.x] += red[threadIdx.x + o]; __syncthreads(); }
    float rstd = rsqrtf(red[0] * (1.f / Dn) + 1e-6f);
    #pragma unroll
    for (int i = 0; i < 4; ++i) {
        int c = threadIdx.x + i * 256;
        float v = (vals[i] - mean) * rstd * sc[c] + bi[c];
        if (ybf) ybf[(size_t)row * Dn + c] = f2bf(v);
        if (yf)  yf [(size_t)row * Dn + c] = v;
    }
}

__global__ __launch_bounds__(256) void softmax_kernel(const float* __restrict__ lg,
                                                      unsigned short* __restrict__ attn) {
    __shared__ float red[256];
    size_t row = blockIdx.x;
    const float* xr = lg + row * Sn;
    float vals[4];
    float mx = -3.4e38f;
    #pragma unroll
    for (int i = 0; i < 4; ++i) { vals[i] = xr[threadIdx.x + i * 256] * 0.125f; mx = fmaxf(mx, vals[i]); }
    red[threadIdx.x] = mx; __syncthreads();
    for (int o = 128; o; o >>= 1) { if (threadIdx.x < o) red[threadIdx.x] = fmaxf(red[threadIdx.x], red[threadIdx.x + o]); __syncthreads(); }
    float m = red[0]; __syncthreads();
    float s = 0.f;
    #pragma unroll
    for (int i = 0; i < 4; ++i) { vals[i] = expf(vals[i] - m); s += vals[i]; }
    red[threadIdx.x] = s; __syncthreads();
    for (int o = 128; o; o >>= 1) { if (threadIdx.x < o) red[threadIdx.x] += red[threadIdx.x + o]; __syncthreads(); }
    float inv = 1.f / red[0];
    #pragma unroll
    for (int i = 0; i < 4; ++i)
        attn[row * Sn + threadIdx.x + i * 256] = f2bf(vals[i] * inv);
}

__global__ __launch_bounds__(256) void act_blend(const float* __restrict__ trans,
                                                 const float* __restrict__ upw,
                                                 float* __restrict__ prev) {
    size_t idx = (size_t)blockIdx.x * 256 + threadIdx.x;
    float w = upw[idx >> 10];
    prev[idx] = trans[idx] * w + prev[idx] * (1.f - w);
}

__global__ __launch_bounds__(256) void write_tails(const float* __restrict__ nupd,
                                                   const float* __restrict__ rem,
                                                   float* __restrict__ out) {
    int i = blockIdx.x * 256 + threadIdx.x;
    if (i < Mn) { out[i] = nupd[i]; out[Mn + i] = rem[i]; }
}

// ---------------- host orchestration ----------------
extern "C" void kernel_launch(void* const* d_in, const int* in_sizes, int n_in,
                              void* d_out, int out_size, void* d_ws, size_t ws_size,
                              hipStream_t stream) {
    (void)in_sizes; (void)n_in; (void)out_size; (void)ws_size;
    const float* x        = (const float*)d_in[0];
    const float* step_emb = (const float*)d_in[1];
    const float* ln1_s = (const float*)d_in[2];
    const float* ln1_b = (const float*)d_in[3];
    const float* ln2_s = (const float*)d_in[4];
    const float* ln2_b = (const float*)d_in[5];
    const float* lnf_s = (const float*)d_in[6];
    const float* lnf_b = (const float*)d_in[7];
    const float* Wq = (const float*)d_in[8];
    const float* Wk = (const float*)d_in[9];
    const float* Wv = (const float*)d_in[10];
    const float* Wo = (const float*)d_in[11];
    const float* W1 = (const float*)d_in[12];
    const float* b1 = (const float*)d_in[13];
    const float* W2 = (const float*)d_in[14];
    const float* b2 = (const float*)d_in[15];
    const float* halt_w = (const float*)d_in[16];
    const float* halt_b = (const float*)d_in[17];

    char* w = (char*)d_ws;
    auto alloc = [&](size_t bytes) {
        void* p = (void*)w;
        w += (bytes + 255) & ~(size_t)255;
        return p;
    };
    unsigned short* Wqkv_t = (unsigned short*)alloc((size_t)3 * Dn * Dn * 2);
    unsigned short* Wo_t   = (unsigned short*)alloc((size_t)Dn * Dn * 2);
    unsigned short* W1_t   = (unsigned short*)alloc((size_t)Fn * Dn * 2);
    unsigned short* W2_t   = (unsigned short*)alloc((size_t)Dn * Fn * 2);
    float* stateA = (float*)alloc((size_t)Mn * Dn * 4);
    float* stateB = (float*)alloc((size_t)Mn * Dn * 4);
    float* prev   = (float*)alloc((size_t)Mn * Dn * 4);
    float* st     = (float*)alloc((size_t)Mn * Dn * 4);
    float* st2    = (float*)alloc((size_t)Mn * Dn * 4);
    unsigned short* ybf   = (unsigned short*)alloc((size_t)Mn * Dn * 2);
    unsigned short* qkvbf = (unsigned short*)alloc((size_t)Mn * 3 * Dn * 2);
    unsigned short* vtbf  = (unsigned short*)alloc((size_t)BHn * DHn * Sn * 2);
    unsigned short* aobf  = (unsigned short*)alloc((size_t)Mn * Dn * 2);
    unsigned short* h1bf  = (unsigned short*)alloc((size_t)Mn * Fn * 2);
    float* halting    = (float*)alloc((size_t)Mn * 4);
    float* remainders = (float*)alloc((size_t)Mn * 4);
    float* nupd       = (float*)alloc((size_t)Mn * 4);
    float* upw        = (float*)alloc((size_t)Mn * 4);
    float* logits = (float*)alloc((size_t)CHn * Sn * Sn * 4);                   // 32 MB
    unsigned short* attnbf = (unsigned short*)alloc((size_t)CHn * Sn * Sn * 2); // 16 MB

    // one-time weight prep (bf16, N-major)
    transpose_w<<<(Dn * Dn) / 256, 256, 0, stream>>>(Wq, Wqkv_t, Dn, Dn);
    transpose_w<<<(Dn * Dn) / 256, 256, 0, stream>>>(Wk, Wqkv_t + (size_t)Dn * Dn, Dn, Dn);
    transpose_w<<<(Dn * Dn) / 256, 256, 0, stream>>>(Wv, Wqkv_t + (size_t)2 * Dn * Dn, Dn, Dn);
    transpose_w<<<(Dn * Dn) / 256, 256, 0, stream>>>(Wo, Wo_t, Dn, Dn);
    transpose_w<<<(Dn * Fn) / 256, 256, 0, stream>>>(W1, W1_t, Dn, Fn);  // dst [Fn, Dn]
    transpose_w<<<(Fn * Dn) / 256, 256, 0, stream>>>(W2, W2_t, Fn, Dn);  // dst [Dn, Fn]

    (void)hipMemsetAsync(halting,    0, (size_t)Mn * 4, stream);
    (void)hipMemsetAsync(remainders, 0, (size_t)Mn * 4, stream);
    (void)hipMemsetAsync(nupd,       0, (size_t)Mn * 4, stream);
    (void)hipMemsetAsync(prev,       0, (size_t)Mn * Dn * 4, stream);

    float* state = stateA;
    float* trans = stateB;
    const int EW = (Mn * Dn) / 256;   // 8192 blocks for 2M-element maps

    for (int step = 0; step < STEPS; ++step) {
        pos_step<<<EW, 256, 0, stream>>>(step == 0 ? x : state, step_emb + (size_t)step * Dn, st);
        act_update<<<Mn / 8, 256, 0, stream>>>(st, halt_w, halt_b, halting, remainders, nupd, upw);
        ln_kernel<<<Mn, 256, 0, stream>>>(st, ln1_s, ln1_b, ybf, nullptr);

        // fused QKV: [2048,1024] x [1024,3072] -> bf16 qkv
        gemm_tiled<<<dim3(3 * Dn / 128, Mn / 128), 256, 0, stream>>>(
            ybf, Dn, Wqkv_t, Dn, Dn,
            nullptr, qkvbf, 3 * Dn, nullptr, nullptr, 0);
        transpose_v<<<(BHn * DHn * Sn) / 256, 256, 0, stream>>>(qkvbf, vtbf);

        for (int c = 0; c < BHn / CHn; ++c) {
            int pb = c * CHn;
            gemm_qk<<<dim3((Sn / 16) * (Sn / 16) / 8, 1, CHn), 256, 0, stream>>>(qkvbf, pb, logits);
            softmax_kernel<<<CHn * Sn, 256, 0, stream>>>(logits, attnbf);
            gemm_av<<<dim3(8, 1, CHn), 256, 0, stream>>>(attnbf, vtbf, pb, aobf);
        }

        // Wo projection + residual(st) -> st2
        gemm_tiled<<<dim3(Dn / 128, Mn / 128), 256, 0, stream>>>(
            aobf, Dn, Wo_t, Dn, Dn,
            st2, nullptr, Dn, nullptr, st, 0);

        ln_kernel<<<Mn, 256, 0, stream>>>(st2, ln2_s, ln2_b, ybf, nullptr);

        // FFN1 + bias + relu -> bf16 h1
        gemm_tiled<<<dim3(Fn / 128, Mn / 128), 256, 0, stream>>>(
            ybf, Dn, W1_t, Dn, Dn,
            nullptr, h1bf, Fn, b1, nullptr, 1);
        // FFN2 + bias + residual(st2) -> trans
        gemm_tiled<<<dim3(Dn / 128, Mn / 128), 256, 0, stream>>>(
            h1bf, Fn, W2_t, Fn, Fn,
            trans, nullptr, Dn, b2, st2, 0);

        act_blend<<<EW, 256, 0, stream>>>(trans, upw, prev);

        float* tmp = state; state = trans; trans = tmp;
    }

    // final LN(prev) -> d_out[0 : M*D], then n_updates / remainders tails
    ln_kernel<<<Mn, 256, 0, stream>>>(prev, lnf_s, lnf_b, nullptr, (float*)d_out);
    write_tails<<<Mn / 256, 256, 0, stream>>>(nupd, remainders, (float*)d_out + (size_t)Mn * Dn);
}
